// Net_65189013619123
// MI455X (gfx1250) — matrix operand order repaired and checked by reading, not compiled, not run on hardware
//
#include <hip/hip_runtime.h>

#define NUM_IN   8192
#define NUM_HID  4096
#define TSTEPS   256
#define PWIDTH   10
#define BETA1    0.9f
#define BETA2    0.8f
#define DECAY1   0.001f
#define DECAY2   0.0005f
#define INC1     0.05f
#define INC2     0.02f
#define LATENCY  5.0f
#define STDP_LEN 513   // 2*TIME+1

typedef __attribute__((ext_vector_type(16))) _Float16 v16h;
typedef __attribute__((ext_vector_type(8)))  _Float16 v8h;
typedef __attribute__((ext_vector_type(8)))  float    v8f;
typedef int async_v4i __attribute__((vector_size(16)));   // matches builtin proto

// --- gfx1250 async global->LDS copy (ASYNCcnt-tracked), with sync fallback ---
#if __has_builtin(__builtin_amdgcn_global_load_async_to_lds_b128) && \
    __has_builtin(__builtin_amdgcn_s_wait_asynccnt)
typedef __attribute__((address_space(3))) async_v4i* lds_dst_t;
typedef __attribute__((address_space(1))) async_v4i* gbl_src_t;
#define MAKE_LDS_DST(p) ((lds_dst_t)(p))
#define MAKE_GBL_SRC(p) ((gbl_src_t)(p))
#define ASYNC_COPY16(src, dst) \
  __builtin_amdgcn_global_load_async_to_lds_b128((src), (dst), 0, 0)
#define WAIT_ASYNC(n) __builtin_amdgcn_s_wait_asynccnt(n)
#define BCHUNK_STEP 1024          // 32 rows * 256 halfs * 2B / 16B per unit
#else
typedef _Float16*       lds_dst_t;
typedef const _Float16* gbl_src_t;
#define MAKE_LDS_DST(p) ((lds_dst_t)(p))
#define MAKE_GBL_SRC(p) ((gbl_src_t)(p))
#define ASYNC_COPY16(src, dst) (*(v8h*)(dst) = *(const v8h*)(src))
#define WAIT_ASYNC(n) ((void)0)
#define BCHUNK_STEP (32 * TSTEPS) // in halfs
#endif

// ---------------------------------------------------------------------------
// Phase 0a: build latency-coded spike raster R[i][t] (f16, t contiguous)
// ---------------------------------------------------------------------------
__global__ void build_raster(const float* __restrict__ image,
                             _Float16* __restrict__ R) {
  int idx = blockIdx.x * blockDim.x + threadIdx.x;       // 8192*256 threads
  if (idx >= NUM_IN * TSTEPS) return;
  int i = idx >> 8;
  int t = idx & (TSTEPS - 1);
  float st = floorf((float)TSTEPS * image[i]);
  float tf = (float)t;
  R[idx] = (tf >= st && tf < st + (float)PWIDTH) ? (_Float16)1.0f : (_Float16)0.0f;
}

// ---------------------------------------------------------------------------
// Phase 0b: convert w1 f32 -> f16
// ---------------------------------------------------------------------------
__global__ void cvt_w1_f16(const float* __restrict__ w1,
                           _Float16* __restrict__ w1h) {
  size_t idx = (size_t)blockIdx.x * blockDim.x + threadIdx.x;
  if (idx >= (size_t)NUM_HID * NUM_IN) return;
  w1h[idx] = (_Float16)w1[idx];
}

// ---------------------------------------------------------------------------
// Phase 0c: transpose w2 [8192][4096] -> w2T [4096][8192] (coalesced reads)
// ---------------------------------------------------------------------------
__global__ void transpose_w2(const float* __restrict__ w2,
                             float* __restrict__ w2T) {
  size_t idx = (size_t)blockIdx.x * blockDim.x + threadIdx.x;
  if (idx >= (size_t)NUM_IN * NUM_HID) return;
  int h = (int)(idx & (NUM_HID - 1));
  int o = (int)(idx >> 12);
  w2T[(size_t)h * NUM_IN + o] = w2[idx];
}

// ---------------------------------------------------------------------------
// Phase 1: WMMA GEMM  S[h][t] = sum_k w1h[h][k] * R[k][t]
//   M=4096, N=256, K=8192.
//   Block = 8 waves; all waves share one 64-col group, each owns a 16-row tile.
//   B chunk (32K x 64N, 4KB) staged in LDS via async DMA, double-buffered.
//   A fragments pipelined one chunk ahead in registers.
//   Stores St transposed: St[t][h]  (coalesced reads in the scan).
// ---------------------------------------------------------------------------
#define KCHUNKS (NUM_IN / 32)
__global__ __launch_bounds__(256) void snn_gemm(
    const _Float16* __restrict__ w1h,   // [4096][8192]
    const _Float16* __restrict__ R,     // [8192][256]
    float* __restrict__ St) {           // [256][4096]
  __shared__ _Float16 Bb[2][32][72];    // 72-half row stride: 16B-aligned, de-banked

  int tid  = threadIdx.x;
  int lane = tid & 31;
  int wave = tid >> 5;
  int rb   = blockIdx.x >> 2;          // 32 row blocks (8 tiles each)
  int cg   = blockIdx.x & 3;           // 4 col groups of 64
  int h0   = (rb * 8 + wave) * 16;
  int t0b  = cg * 64;
  int m    = lane & 15;
  int half = lane >> 4;

  // async-copy mapping: thread -> one 16B transfer of the 32x64 chunk.
  // Address-space casts hoisted out of the loop (avoid per-iter aperture code).
  int ck = tid >> 3;                   // chunk row k (0..31)
  int cc = tid & 7;                    // 16B column slot (0..7)
  gbl_src_t gsrc = MAKE_GBL_SRC(R + (size_t)ck * TSTEPS + t0b + cc * 8);
  lds_dst_t bdst[2] = { MAKE_LDS_DST(&Bb[0][ck][cc * 8]),
                        MAKE_LDS_DST(&Bb[1][ck][cc * 8]) };

  const _Float16* abase = w1h + (size_t)(h0 + m) * NUM_IN + half * 8;

  // prologue: DMA chunk 0, prefetch A chunk 0 into registers
  ASYNC_COPY16(gsrc, bdst[0]);
  v8h alo = *(const v8h*)(abase);
  v8h ahi = *(const v8h*)(abase + 16);

  v8f acc[4] = {};
  for (int kc = 0; kc < KCHUNKS; ++kc) {
    int cur = kc & 1;
    if (kc + 1 < KCHUNKS) {
      ASYNC_COPY16(gsrc + (size_t)(kc + 1) * BCHUNK_STEP, bdst[cur ^ 1]);
      WAIT_ASYNC(1);                   // in-order: chunk kc has landed
    } else {
      WAIT_ASYNC(0);
    }
    __syncthreads();

    // A fragment for this chunk (ISA 16-bit 16x32 layout)
    v16h a;
#pragma unroll
    for (int e = 0; e < 8; e++) { a[e] = alo[e]; a[e + 8] = ahi[e]; }

    // pipeline: start loading next A chunk while WMMAs run
    if (kc + 1 < KCHUNKS) {
      const _Float16* an = abase + (size_t)(kc + 1) * 32;
      alo = *(const v8h*)(an);
      ahi = *(const v8h*)(an + 16);
      __builtin_prefetch((const void*)(an + 128), 0, 3);   // global_prefetch_b8
    }

    // B fragments from LDS: lane%16 = N, element = K (table order, matches A).
    // Load all four first so the 64 16-bit LDS reads batch ahead of the WMMAs.
    const _Float16* brow = &Bb[cur][0][m];
    v16h bfrag[4];
#pragma unroll
    for (int ct = 0; ct < 4; ct++)
#pragma unroll
      for (int s = 0; s < 16; s++)
        bfrag[ct][s] = brow[((s & 8) * 2 + 8 * half + (s & 7)) * 72 + ct * 16];
#pragma unroll
    for (int ct = 0; ct < 4; ct++)
      acc[ct] = __builtin_amdgcn_wmma_f32_16x16x32_f16(
          false, a, false, bfrag[ct], (short)0, acc[ct], false, false);
    __syncthreads();                   // buf[cur] free for refill next iter
  }

  // D layout: VGPR v -> M = v + 8*half ; N = lane&15
  int hb = h0 + 8 * half;
#pragma unroll
  for (int ct = 0; ct < 4; ct++) {
    int t = t0b + ct * 16 + m;
#pragma unroll
    for (int v = 0; v < 8; v++)
      St[(size_t)t * NUM_HID + hb + v] = acc[ct][v];
  }
}

// ---------------------------------------------------------------------------
// Phase 2: sequential LIF scan — one workgroup, 1024 threads, all state in regs
// ---------------------------------------------------------------------------
__global__ __launch_bounds__(1024) void snn_scan(
    const float* __restrict__ St,      // [256][4096]
    const float* __restrict__ w2T,     // [4096][8192]
    const float* __restrict__ thresh1,
    const float* __restrict__ thresh2,
    const float* __restrict__ image,
    float* __restrict__ r1f, float* __restrict__ maskf,
    float* __restrict__ relevance,
    float* __restrict__ r2f, float* __restrict__ rho,
    float* __restrict__ errOut /* d_out: [0..8191]=error, [8192]=scalar */) {
  __shared__ float s_val[1024];
  __shared__ int   s_idx[1024];
  __shared__ float s_fire;
  __shared__ int   s_win;

  int tid = threadIdx.x;
  float mem1[4], thr1[4], h1v[4], r1v[4], actv[4];
#pragma unroll
  for (int j = 0; j < 4; j++) {
    thr1[j] = thresh1[tid + j * 1024];
    mem1[j] = h1v[j] = r1v[j] = actv[j] = 0.0f;
  }
  float mem2[8], thr2[8], h2v[8], r2v[8];
#pragma unroll
  for (int j = 0; j < 8; j++) {
    thr2[j] = thresh2[tid + j * 1024];
    mem2[j] = h2v[j] = r2v[j] = 0.0f;
  }

  for (int t = 0; t < TSTEPS; t++) {
    // layer-1 integrate + per-thread argmax
    float bv = -3.4e38f; int bi = 0;
#pragma unroll
    for (int j = 0; j < 4; j++) {
      int h = tid + j * 1024;
      float v = BETA1 * mem1[j] + St[(size_t)t * NUM_HID + h];
      mem1[j] = v;
      if (v > bv) { bv = v; bi = h; }
    }
    s_val[tid] = bv; s_idx[tid] = bi;
    __syncthreads();
    for (int s = 512; s > 0; s >>= 1) {
      if (tid < s) {
        float ov = s_val[tid + s]; int oi = s_idx[tid + s];
        if (ov > s_val[tid] || (ov == s_val[tid] && oi < s_idx[tid])) {
          s_val[tid] = ov; s_idx[tid] = oi;
        }
      }
      __syncthreads();
    }
    int   w  = s_idx[0];
    float mv = s_val[0];

    // winner-owning thread decides fire and updates winner state
    if (tid == (w & 1023)) {
      int j = w >> 10;
      float fire = (mv > thr1[j]) ? 1.0f : 0.0f;
      s_fire = fire; s_win = w;
      mem1[j] *= (1.0f - fire);
      actv[j] += fire;
      if (fire > 0.0f) {
        float tt = (float)t;
        h1v[j] = fmaxf(h1v[j], tt);
        if (r1v[j] == 0.0f) r1v[j] = h1v[j];
      }
      thr1[j] += INC1 * fire;
    }
    __syncthreads();
    float fire = s_fire;
    int   win  = s_win;
#pragma unroll
    for (int j = 0; j < 4; j++) thr1[j] -= DECAY1;

    // layer-2 LIF (reset-by-subtraction) driven by winner column of w2
    const float* col = w2T + (size_t)win * NUM_IN;
#pragma unroll
    for (int j = 0; j < 8; j++) {
      int o = tid + j * 1024;
      float x2 = (fire > 0.0f) ? col[o] : 0.0f;
      float v  = BETA2 * mem2[j] + x2;
      float s2 = (v > thr2[j]) ? 1.0f : 0.0f;
      v -= s2 * thr2[j];
      mem2[j] = v;
      thr2[j] += -DECAY2 + INC2 * s2;
      if (s2 > 0.0f) {
        float tt = (float)t;
        h2v[j] = fmaxf(h2v[j], tt);
        if (r2v[j] == 0.0f) r2v[j] = h2v[j];
      }
    }
    __syncthreads();   // protect s_val/s_idx/s_fire reuse
  }

  // epilogue: first-spike times, masks, relevance, rho, error (+scalar)
  const float tf = (float)TSTEPS;
#pragma unroll
  for (int j = 0; j < 4; j++) {
    int h = tid + j * 1024;
    float r = (r1v[j] == 0.0f) ? tf : r1v[j];
    float mk = (actv[j] > 0.0f) ? 1.0f : 0.0f;
    r1f[h] = r;
    maskf[h] = mk;
    relevance[h] = (tf - r) * mk / tf;
  }
  float psum = 0.0f;
#pragma unroll
  for (int j = 0; j < 8; j++) {
    int o = tid + j * 1024;
    float r = (r2v[j] == 0.0f) ? tf : r2v[j];
    r2f[o] = r;
    float it = tf * image[o];
    rho[o] = (r - LATENCY - it) / tf + 0.15f;
    float e = (it - (r - LATENCY)) / tf;
    errOut[o] = e;
    psum += e * e;
  }
  s_val[tid] = psum;
  __syncthreads();
  for (int s = 512; s > 0; s >>= 1) {
    if (tid < s) s_val[tid] += s_val[tid + s];
    __syncthreads();
  }
  if (tid == 0) errOut[NUM_IN] = s_val[0];
}

// ---------------------------------------------------------------------------
// Phase 3a: STDP update for w1  (out[h][i])
// ---------------------------------------------------------------------------
__global__ void w1_update(const float* __restrict__ w1,
                          const float* __restrict__ stdp1,
                          const float* __restrict__ image,
                          const float* __restrict__ r1f,
                          const float* __restrict__ maskf,
                          float* __restrict__ out) {
  size_t idx = (size_t)blockIdx.x * blockDim.x + threadIdx.x;
  if (idx >= (size_t)NUM_HID * NUM_IN) return;
  int h = (int)(idx >> 13);
  int i = (int)(idx & (NUM_IN - 1));
  float it = (float)TSTEPS * image[i];
  float delta = (r1f[h] - it) * maskf[h];
  int k = (int)(delta + (float)TSTEPS);   // + OFFSET - 1
  k = min(max(k, 0), STDP_LEN - 1);
  float v = stdp1[k] + w1[idx];
  out[idx] = fminf(fmaxf(v, 0.0f), 1.0f);
}

// ---------------------------------------------------------------------------
// Phase 3b: STDP update for w2  (out[o][h])
// ---------------------------------------------------------------------------
__global__ void w2_update(const float* __restrict__ w2,
                          const float* __restrict__ stdp2,
                          const float* __restrict__ r1f,
                          const float* __restrict__ r2f,
                          const float* __restrict__ rho,
                          const float* __restrict__ relevance,
                          float* __restrict__ out) {
  size_t idx = (size_t)blockIdx.x * blockDim.x + threadIdx.x;
  if (idx >= (size_t)NUM_IN * NUM_HID) return;
  int h = (int)(idx & (NUM_HID - 1));
  int o = (int)(idx >> 12);
  float delta = r2f[o] - r1f[h];
  int k = (int)(delta + (float)TSTEPS);
  k = min(max(k, 0), STDP_LEN - 1);
  float dw = stdp2[k] * (rho[o] * relevance[h]);
  float v = dw + w2[idx];
  out[idx] = fminf(fmaxf(v, 0.0f), 1.0f);
}

// ---------------------------------------------------------------------------
extern "C" void kernel_launch(void* const* d_in, const int* in_sizes, int n_in,
                              void* d_out, int out_size, void* d_ws, size_t ws_size,
                              hipStream_t stream) {
  const float* image   = (const float*)d_in[0];
  const float* w1      = (const float*)d_in[1];
  const float* w2      = (const float*)d_in[2];
  const float* thresh1 = (const float*)d_in[3];
  const float* thresh2 = (const float*)d_in[4];
  const float* stdp1   = (const float*)d_in[5];
  const float* stdp2   = (const float*)d_in[6];
  (void)in_sizes; (void)n_in; (void)out_size; (void)ws_size;

  // scratch carve-out
  char* ws = (char*)d_ws;
  _Float16* w1h = (_Float16*)(ws);                                  // 64 MB
  _Float16* R   = (_Float16*)(ws + (size_t)67108864);               // 4 MB
  float*    St  = (float*)   (ws + (size_t)71303168);               // 4 MB
  float*    w2T = (float*)   (ws + (size_t)75497472);               // 128 MB
  float*    r1f = (float*)   (ws + (size_t)209715200);
  float*    mkf = (float*)   (ws + (size_t)209715200 + 16384);
  float*    rel = (float*)   (ws + (size_t)209715200 + 32768);
  float*    r2f = (float*)   (ws + (size_t)209715200 + 49152);
  float*    rho = (float*)   (ws + (size_t)209715200 + 81920);

  float* out      = (float*)d_out;            // [0..8191] error, [8192] scalar
  float* w1_out   = out + 8193;               // 4096*8192
  float* w2_out   = out + 8193 + (size_t)NUM_HID * NUM_IN;

  const size_t NW1 = (size_t)NUM_HID * NUM_IN;  // 33554432

  build_raster<<<(NUM_IN * TSTEPS + 255) / 256, 256, 0, stream>>>(image, R);
  cvt_w1_f16<<<(unsigned)((NW1 + 255) / 256), 256, 0, stream>>>(w1, w1h);
  transpose_w2<<<(unsigned)((NW1 + 255) / 256), 256, 0, stream>>>(w2, w2T);

  // 1024 waves total: (4096/16) row tiles * (256/64) col groups, 8 waves/block
  snn_gemm<<<128, 256, 0, stream>>>(w1h, R, St);

  snn_scan<<<1, 1024, 0, stream>>>(St, w2T, thresh1, thresh2, image,
                                   r1f, mkf, rel, r2f, rho, out);

  w1_update<<<(unsigned)((NW1 + 255) / 256), 256, 0, stream>>>(
      w1, stdp1, image, r1f, mkf, w1_out);
  w2_update<<<(unsigned)((NW1 + 255) / 256), 256, 0, stream>>>(
      w2, stdp2, r1f, r2f, rho, rel, w2_out);
}
